// VAE_50620484551196
// MI455X (gfx1250) — compile-verified, matrix operand order checked by reading, not run on hardware
//
#include <hip/hip_runtime.h>
#include <stdint.h>

// ---------------------------------------------------------------------------
// Types for CDNA5 WMMA (gfx1250, wave32)
// ---------------------------------------------------------------------------
typedef __bf16 bf16;
typedef __attribute__((ext_vector_type(16))) __bf16 v16bf;
typedef __attribute__((ext_vector_type(8)))  __bf16 v8bf;
typedef __attribute__((ext_vector_type(8)))  float  v8f;
typedef __attribute__((ext_vector_type(4)))  int    v4i;

// Problem constants
#define BB   4096
#define DD   4624
#define DPAD 4672   // 4624 padded to multiple of 64 (73*64) for BK=64 tiles
#define LL   68
#define EE   1024

#if __has_builtin(__builtin_amdgcn_global_load_async_to_lds_b128)
#define HAVE_ASYNC 1
#else
#define HAVE_ASYNC 0
#endif

#if HAVE_ASYNC
__device__ __forceinline__ void async_cp16(const bf16* g, bf16* l) {
  // GLOBAL_LOAD_ASYNC_TO_LDS_B128: memory -> LDS, tracked by ASYNCcnt.
  __builtin_amdgcn_global_load_async_to_lds_b128((v4i*)g, (v4i*)l, 0, 0);
}
#endif

__device__ __forceinline__ void wait_async_le8() {
#if __has_builtin(__builtin_amdgcn_s_wait_asynccnt)
  __builtin_amdgcn_s_wait_asynccnt(8);
#else
  asm volatile("s_wait_asynccnt 0x8" ::: "memory");
#endif
}
__device__ __forceinline__ void wait_async_le0() {
#if __has_builtin(__builtin_amdgcn_s_wait_asynccnt)
  __builtin_amdgcn_s_wait_asynccnt(0);
#else
  asm volatile("s_wait_asynccnt 0x0" ::: "memory");
#endif
}

// ---------------------------------------------------------------------------
// Pack: f32 row-major [rows,K] -> bf16 [rows,Kpad], zero-padding cols K..Kpad
// ---------------------------------------------------------------------------
__global__ __launch_bounds__(256) void pack_bf16(const float* __restrict__ src,
                                                 bf16* __restrict__ dst,
                                                 int rows, int K, int Kpad) {
  int i = blockIdx.x * blockDim.x + threadIdx.x;
  int total = rows * Kpad;
  if (i >= total) return;
  int r = i / Kpad;
  int c = i - r * Kpad;
  dst[i] = (c < K) ? (bf16)src[(size_t)r * K + c] : (bf16)0.0f;
}

__global__ __launch_bounds__(256) void copy_f32(const float* __restrict__ src,
                                                float* __restrict__ dst, int n) {
  int i = blockIdx.x * blockDim.x + threadIdx.x;
  if (i < n) dst[i] = src[i];
}

// ---------------------------------------------------------------------------
// WMMA fragment load from LDS row (ISA 16-bit A/B layout):
//   lanes 0-15 : K = {0..7, 16..23};  lanes 16-31: K = {8..15, 24..31}
// Two 16-byte ds loads per fragment.
// ---------------------------------------------------------------------------
__device__ __forceinline__ v16bf load_frag(const bf16* row, int kh) {
  v8bf lo = *(const v8bf*)(row + kh);
  v8bf hi = *(const v8bf*)(row + 16 + kh);
  v16bf r;
#pragma unroll
  for (int i = 0; i < 8; ++i) { r[i] = lo[i]; r[i + 8] = hi[i]; }
  return r;
}

// ---------------------------------------------------------------------------
// Tiled bf16 WMMA GEMM:  C[M,N] = act( A[M,K] * B[N,K]^T + bias[N] )
//   Block tile 128x128x64, 256 threads (8 waves: 2 in M x 4 in N),
//   each wave: 4 M-tiles x 2 N-tiles -> v_wmma_f32_16x16x32_bf16, f32 acc,
//   2 K-substeps per LDS tile. Double-buffered LDS fed by
//   GLOBAL_LOAD_ASYNC_TO_LDS_B128 (ASYNCcnt pipeline, 8 ops in flight per
//   thread while computing); sync load+ds_store fallback otherwise.
// ACT: 0 = none (f32 out), 1 = ReLU (bf16 out), 2 = sigmoid (f32 out)
// Requires: M % 128 == 0, K % 64 == 0 with K/64 >= 2, 16B-aligned rows.
// ---------------------------------------------------------------------------
template <int ACT>
__global__ __launch_bounds__(256) void gemm_wmma_bf16(
    const bf16* __restrict__ A, const bf16* __restrict__ Bm,
    const float* __restrict__ bias, void* __restrict__ Cout,
    int M, int Nn, int K, int lda, int ldb, int ldc) {
  __shared__ bf16 sA[2][128 * 72];  // 64 cols padded to 72 (144B rows, 16B aligned)
  __shared__ bf16 sB[2][128 * 72];

  const int tid  = threadIdx.x;
  const int wave = tid >> 5;
  const int lane = tid & 31;
  const int wy   = wave >> 2;      // 0..1  (M direction, 64 rows each)
  const int wx   = wave & 3;       // 0..3  (N direction, 32 cols each)
  const int lrow = lane & 15;
  const int half = lane >> 4;      // 0 or 1
  const int kh   = half * 8;

  const int mBase = blockIdx.y * 128;
  const int nBase = blockIdx.x * 128;

  // Per-thread staging coordinates: 4 x 16B chunks per matrix per tile
  // (128 rows x 64 cols bf16 = 16KB = 256 threads x 4 x 16B).
  int rr[4], cc[4], nr[4];
#pragma unroll
  for (int j = 0; j < 4; ++j) {
    int lin = tid + j * 256;
    rr[j] = lin >> 3;            // 0..127
    cc[j] = (lin & 7) * 8;       // 0,8,...,56 (bf16 elems)
    // Clamp B rows instead of zero-filling: out-of-range output columns are
    // never stored, and column n of D only reads row n of B.
    nr[j] = (nBase + rr[j] < Nn) ? (nBase + rr[j]) : (Nn - 1);
  }

  v8f acc[4][2];
#pragma unroll
  for (int mt = 0; mt < 4; ++mt)
#pragma unroll
    for (int nt = 0; nt < 2; ++nt)
#pragma unroll
      for (int r = 0; r < 8; ++r) acc[mt][nt][r] = 0.0f;

  auto compute = [&](int buf) {
#pragma unroll
    for (int ks = 0; ks < 2; ++ks) {  // two 32-deep substeps per 64-deep tile
      v16bf af[4], bfr[2];
#pragma unroll
      for (int mt = 0; mt < 4; ++mt)
        af[mt] = load_frag(&sA[buf][(wy * 64 + mt * 16 + lrow) * 72 + ks * 32], kh);
#pragma unroll
      for (int nt = 0; nt < 2; ++nt)
        bfr[nt] = load_frag(&sB[buf][(wx * 32 + nt * 16 + lrow) * 72 + ks * 32], kh);
#pragma unroll
      for (int mt = 0; mt < 4; ++mt)
#pragma unroll
        for (int nt = 0; nt < 2; ++nt)
          acc[mt][nt] = __builtin_amdgcn_wmma_f32_16x16x32_bf16(
              false, af[mt], false, bfr[nt], (short)0, acc[mt][nt], false, false);
    }
  };

#if HAVE_ASYNC
  auto issue = [&](int k0, int buf) {
#pragma unroll
    for (int j = 0; j < 4; ++j)
      async_cp16(&A[(size_t)(mBase + rr[j]) * lda + k0 + cc[j]],
                 &sA[buf][rr[j] * 72 + cc[j]]);
#pragma unroll
    for (int j = 0; j < 4; ++j)
      async_cp16(&Bm[(size_t)nr[j] * ldb + k0 + cc[j]],
                 &sB[buf][rr[j] * 72 + cc[j]]);
  };
  const int nK = K >> 6;
  issue(0, 0);
  for (int i = 0; i < nK - 1; ++i) {       // steady state: no conditionals
    issue((i + 1) << 6, (i + 1) & 1);      // overlap next tile with compute
    wait_async_le8();                      // tile i's 8 ops complete (in-order)
    __syncthreads();                       // all waves' tile-i data visible
    compute(i & 1);
    __syncthreads();                       // LDS buffer free for reuse
  }
  wait_async_le0();                        // pipeline tail
  __syncthreads();
  compute((nK - 1) & 1);
#else
  for (int k0 = 0; k0 < K; k0 += 64) {
    float4 va[4], vb[4];
#pragma unroll
    for (int j = 0; j < 4; ++j) {
      va[j] = *(const float4*)(&A[(size_t)(mBase + rr[j]) * lda + k0 + cc[j]]);
      vb[j] = *(const float4*)(&Bm[(size_t)nr[j] * ldb + k0 + cc[j]]);
    }
#pragma unroll
    for (int j = 0; j < 4; ++j) {
      *(float4*)(&sA[0][rr[j] * 72 + cc[j]]) = va[j];
      *(float4*)(&sB[0][rr[j] * 72 + cc[j]]) = vb[j];
    }
    __syncthreads();
    compute(0);
    __syncthreads();
  }
#endif

  // ---- epilogue: D-layout VGPR r -> (M = tile + r + half*8, N = tile + lrow)
#pragma unroll
  for (int mt = 0; mt < 4; ++mt) {
#pragma unroll
    for (int nt = 0; nt < 2; ++nt) {
      int n = nBase + wx * 32 + nt * 16 + lrow;
      if (n < Nn) {
        float bv = bias[n];
#pragma unroll
        for (int r = 0; r < 8; ++r) {
          int m = mBase + wy * 64 + mt * 16 + half * 8 + r;
          float v = acc[mt][nt][r] + bv;
          if (ACT == 1) {
            v = v > 0.0f ? v : 0.0f;
            ((bf16*)Cout)[(size_t)m * ldc + n] = (bf16)v;
          } else if (ACT == 2) {
            v = 1.0f / (1.0f + __expf(-v));
            ((float*)Cout)[(size_t)m * ldc + n] = v;
          } else {
            ((float*)Cout)[(size_t)m * ldc + n] = v;
          }
        }
      }
    }
  }
}

// ---------------------------------------------------------------------------
// Latent: z = mu + eps*exp(0.5*logvar);  y = mu . fcy_w + fcy_b
// ---------------------------------------------------------------------------
__global__ __launch_bounds__(256) void zy_kernel(
    const float* __restrict__ mu, const float* __restrict__ logvar,
    const float* __restrict__ eps, const float* __restrict__ fcy_w,
    const float* __restrict__ fcy_b, float* __restrict__ z,
    float* __restrict__ y) {
  int b = blockIdx.x * blockDim.x + threadIdx.x;
  if (b >= BB) return;
  float yv = fcy_b[0];
#pragma unroll 4
  for (int l = 0; l < LL; ++l) {
    float m  = mu[b * LL + l];
    float lv = logvar[b * LL + l];
    z[b * LL + l] = m + eps[b * LL + l] * __expf(0.5f * lv);
    yv += m * fcy_w[l];
  }
  y[b] = yv;
}

// out[b,o] = act( sum_d in[b,d] * W[o,d] + bias[o] ),  W row-major [68,68]
__global__ __launch_bounds__(256) void branch_kernel(
    const float* __restrict__ in, const float* __restrict__ W,
    const float* __restrict__ bias, float* __restrict__ out, int sig) {
  int i = blockIdx.x * blockDim.x + threadIdx.x;
  if (i >= BB * LL) return;
  int b = i / LL;
  int o = i - b * LL;
  float acc = bias[o];
  const float* ir = in + b * LL;
  const float* wr = W + o * LL;
#pragma unroll 4
  for (int d = 0; d < LL; ++d) acc += ir[d] * wr[d];
  if (sig) acc = 1.0f / (1.0f + __expf(-acc));
  out[i] = acc;
}

// outer[b, i*68+j] = h51[b,i] * h51[b,j], stored bf16 with K padded to DPAD
__global__ __launch_bounds__(256) void outer_kernel(
    const float* __restrict__ h51, bf16* __restrict__ ob) {
  long long i = (long long)blockIdx.x * blockDim.x + threadIdx.x;
  if (i >= (long long)BB * DPAD) return;
  int b = (int)(i / DPAD);
  int c = (int)(i - (long long)b * DPAD);
  float v = 0.0f;
  if (c < DD) {
    int r = c / LL;
    int j = c - r * LL;
    v = h51[b * LL + r] * h51[b * LL + j];
  }
  ob[i] = (bf16)v;
}

// ---------------------------------------------------------------------------
// Orchestration
// ---------------------------------------------------------------------------
extern "C" void kernel_launch(void* const* d_in, const int* in_sizes, int n_in,
                              void* d_out, int out_size, void* d_ws, size_t ws_size,
                              hipStream_t stream) {
  const float* x      = (const float*)d_in[0];
  const float* eps    = (const float*)d_in[1];
  const float* fc11_w = (const float*)d_in[2];
  const float* fc11_b = (const float*)d_in[3];
  const float* fc12_w = (const float*)d_in[4];
  const float* fc12_b = (const float*)d_in[5];
  const float* fc21_w = (const float*)d_in[6];
  const float* fc21_b = (const float*)d_in[7];
  const float* fc22_w = (const float*)d_in[8];
  const float* fc22_b = (const float*)d_in[9];
  const float* W3     = (const float*)d_in[10];  // [5,68,68], branch 0 at offset 0
  const float* b3     = (const float*)d_in[11];
  const float* W4     = (const float*)d_in[12];
  const float* b4     = (const float*)d_in[13];
  const float* W5     = (const float*)d_in[14];  // idx[0] == 0
  const float* b5     = (const float*)d_in[15];
  const float* fc6_w  = (const float*)d_in[16];
  const float* fc6_b  = (const float*)d_in[17];
  const float* fcy_w  = (const float*)d_in[18];
  const float* fcy_b  = (const float*)d_in[19];

  // Output layout: recon [B,D] | mu [B,L] | logvar [B,L] | y [B]
  float* out_recon  = (float*)d_out;
  float* out_mu     = out_recon + (size_t)BB * DD;
  float* out_logvar = out_mu + (size_t)BB * LL;
  float* out_y      = out_logvar + (size_t)BB * LL;

  // Workspace layout (256B-aligned chunks)
  char* ws = (char*)d_ws;
  size_t off = 0;
  auto take = [&](size_t bytes) {
    char* p = ws + off;
    off += (bytes + 255) & ~(size_t)255;
    return p;
  };
  bf16*  xb   = (bf16*)take((size_t)BB * DPAD * 2);   // x bf16; later reused for outer
  bf16*  w1b  = (bf16*)take((size_t)2 * EE * DPAD * 2);
  float* b1   = (float*)take((size_t)2 * EE * 4);
  bf16*  hb   = (bf16*)take((size_t)BB * 2 * EE * 2);
  bf16*  w21b = (bf16*)take((size_t)LL * EE * 2);
  bf16*  w22b = (bf16*)take((size_t)LL * EE * 2);
  bf16*  w6b  = (bf16*)take((size_t)DD * DPAD * 2);
  float* zbuf = (float*)take((size_t)BB * LL * 4);
  float* h3b  = (float*)take((size_t)BB * LL * 4);
  float* h4b  = (float*)take((size_t)BB * LL * 4);
  float* h51  = (float*)take((size_t)BB * LL * 4);
  (void)ws_size; (void)in_sizes; (void)n_in; (void)out_size;

  auto cdiv = [](long long a, long long b) { return (int)((a + b - 1) / b); };

  // ---- 1. pack operands to bf16 (K padded to 4672 where K = 4624) ----
  pack_bf16<<<cdiv((long long)BB * DPAD, 256), 256, 0, stream>>>(x, xb, BB, DD, DPAD);
  pack_bf16<<<cdiv((long long)EE * DPAD, 256), 256, 0, stream>>>(fc11_w, w1b, EE, DD, DPAD);
  pack_bf16<<<cdiv((long long)EE * DPAD, 256), 256, 0, stream>>>(fc12_w, w1b + (size_t)EE * DPAD, EE, DD, DPAD);
  copy_f32<<<cdiv(EE, 256), 256, 0, stream>>>(fc11_b, b1, EE);
  copy_f32<<<cdiv(EE, 256), 256, 0, stream>>>(fc12_b, b1 + EE, EE);
  pack_bf16<<<cdiv((long long)LL * EE, 256), 256, 0, stream>>>(fc21_w, w21b, LL, EE, EE);
  pack_bf16<<<cdiv((long long)LL * EE, 256), 256, 0, stream>>>(fc22_w, w22b, LL, EE, EE);
  pack_bf16<<<cdiv((long long)DD * DPAD, 256), 256, 0, stream>>>(fc6_w, w6b, DD, DD, DPAD);

  // ---- 2. encoder GEMM: h = relu(x @ [fc11;fc12]^T + b), bf16 out [B,2048] ----
  {
    dim3 grid(cdiv(2 * EE, 128), BB / 128);
    gemm_wmma_bf16<1><<<grid, 256, 0, stream>>>(xb, w1b, b1, hb,
                                                BB, 2 * EE, DPAD, DPAD, DPAD, 2 * EE);
  }

  // ---- 3. mu / logvar GEMMs (N=68, K=1024), f32 straight into d_out ----
  {
    dim3 grid(cdiv(LL, 128), BB / 128);
    gemm_wmma_bf16<0><<<grid, 256, 0, stream>>>(hb, w21b, fc21_b, out_mu,
                                                BB, LL, EE, 2 * EE, EE, LL);
    gemm_wmma_bf16<0><<<grid, 256, 0, stream>>>(hb + EE, w22b, fc22_b, out_logvar,
                                                BB, LL, EE, 2 * EE, EE, LL);
  }

  // ---- 4. latent chain (only branch 0 is live in the reference) ----
  zy_kernel<<<cdiv(BB, 256), 256, 0, stream>>>(out_mu, out_logvar, eps, fcy_w, fcy_b,
                                               zbuf, out_y);
  branch_kernel<<<cdiv((long long)BB * LL, 256), 256, 0, stream>>>(zbuf, W3, b3, h3b, 0);
  branch_kernel<<<cdiv((long long)BB * LL, 256), 256, 0, stream>>>(h3b, W4, b4, h4b, 1);
  branch_kernel<<<cdiv((long long)BB * LL, 256), 256, 0, stream>>>(h4b, W5, b5, h51, 0);

  // ---- 5. outer product -> bf16 [B, 4672] (reuse xb region) ----
  outer_kernel<<<cdiv((long long)BB * DPAD, 256), 256, 0, stream>>>(h51, xb);

  // ---- 6. recon GEMM: sigmoid(outer @ fc6_w^T + fc6_b), f32 -> d_out ----
  {
    dim3 grid(cdiv(DD, 128), BB / 128);
    gemm_wmma_bf16<2><<<grid, 256, 0, stream>>>(xb, w6b, fc6_b, out_recon,
                                                BB, DD, DPAD, DPAD, DPAD, DD);
  }
}
